// KNNAttention_16587163697314
// MI455X (gfx1250) — matrix operand order repaired and checked by reading, not compiled, not run on hardware
//
#include <hip/hip_runtime.h>
#include <hip/hip_bf16.h>

typedef __attribute__((ext_vector_type(16))) _Float16 v16h;
typedef __attribute__((ext_vector_type(8)))  _Float16 v8h;
typedef __attribute__((ext_vector_type(4)))  _Float16 v4h;
typedef __attribute__((ext_vector_type(8)))  float    v8f;

__device__ __forceinline__ v16h cat8(v8h lo, v8h hi) {
    return __builtin_shufflevector(lo, hi, 0, 1, 2, 3, 4, 5, 6, 7,
                                   8, 9, 10, 11, 12, 13, 14, 15);
}

#define NEG_MAX (-3.402823466e38f)

// ---------------------------------------------------------------------------
// C[M,N] = A[M,K] * B[K,N], all f32 in memory, f16 WMMA with f32 accumulate.
// Block: 256 threads (8 waves). Block tile 128x128, K-step 32.
// Waves: 4 along M x 2 along N -> wave tile 32x64 (2x4 C fragments, 8 WMMA/step).
// Register double-buffering: tile k+1 global loads issued before tile k compute.
// Requires M%128==0, N%128==0, K%32==0.
// ---------------------------------------------------------------------------
__global__ __launch_bounds__(256) void gemm_f16_wmma(
    const float* __restrict__ A, const float* __restrict__ Bm,
    float* __restrict__ C, int M, int N, int K) {
    __shared__ _Float16 As[128][40];   // [m][k]
    __shared__ _Float16 Bs[128][40];   // [n][k] (transposed stage)

    const int tid  = threadIdx.x;
    const int lane = tid & 31;
    const int wave = tid >> 5;
    const int wm   = wave & 3;         // 4 waves along M, 32 rows each
    const int wn   = wave >> 2;        // 2 waves along N, 64 cols each
    const int ln16 = lane & 15;
    const int hi16 = lane >> 4;
    const int m0 = blockIdx.x * 128;
    const int n0 = blockIdx.y * 128;

    // per-thread staging coordinates (1024 float4 per tile, 4 per thread)
    const int ar[4]  = {(tid + 0) >> 3, (tid + 256) >> 3, (tid + 512) >> 3, (tid + 768) >> 3};
    const int ac4    = (tid & 7) * 4;                 // A: 8 float4 per 32-col row
    const int bk4[4] = {(tid + 0) >> 5, (tid + 256) >> 5, (tid + 512) >> 5, (tid + 768) >> 5};
    const int bn4    = (tid & 31) * 4;                // B: 32 float4 per 128-col row

    v8f acc[2][4];
#pragma unroll
    for (int a = 0; a < 2; ++a)
#pragma unroll
        for (int f = 0; f < 4; ++f)
#pragma unroll
            for (int i = 0; i < 8; ++i) acc[a][f][i] = 0.0f;

    float4 ra[4], rb[4];
    // preload tile k0 = 0
#pragma unroll
    for (int i = 0; i < 4; ++i) {
        ra[i] = *(const float4*)(A + (size_t)(m0 + ar[i]) * K + ac4);
        rb[i] = *(const float4*)(Bm + (size_t)bk4[i] * N + n0 + bn4);
    }

    for (int k0 = 0; k0 < K; k0 += 32) {
        // store staged registers to LDS (f32 -> f16)
#pragma unroll
        for (int i = 0; i < 4; ++i) {
            v4h p = {(_Float16)ra[i].x, (_Float16)ra[i].y,
                     (_Float16)ra[i].z, (_Float16)ra[i].w};
            *(v4h*)&As[ar[i]][ac4] = p;
            Bs[bn4 + 0][bk4[i]] = (_Float16)rb[i].x;
            Bs[bn4 + 1][bk4[i]] = (_Float16)rb[i].y;
            Bs[bn4 + 2][bk4[i]] = (_Float16)rb[i].z;
            Bs[bn4 + 3][bk4[i]] = (_Float16)rb[i].w;
        }
        __syncthreads();

        // issue next tile's global loads before the compute (latency hiding)
        const int k1 = k0 + 32;
        if (k1 < K) {
#pragma unroll
            for (int i = 0; i < 4; ++i) {
                ra[i] = *(const float4*)(A + (size_t)(m0 + ar[i]) * K + k1 + ac4);
                rb[i] = *(const float4*)(Bm + (size_t)(k1 + bk4[i]) * N + n0 + bn4);
            }
        }

        // fragments + 8 WMMA
        const int hh = hi16 * 8;
        const int kb = hi16 * 16;
        v16h af[2];
#pragma unroll
        for (int a = 0; a < 2; ++a) {
            int arow = wm * 32 + a * 16 + ln16;
            af[a] = cat8(*(const v8h*)&As[arow][hh],
                         *(const v8h*)&As[arow][16 + hh]);
        }
#pragma unroll
        for (int f = 0; f < 4; ++f) {
            int bcol = wn * 64 + f * 16 + ln16;
            v16h bf = cat8(*(const v8h*)&Bs[bcol][hh],
                           *(const v8h*)&Bs[bcol][16 + hh]);
#pragma unroll
            for (int a = 0; a < 2; ++a)
                acc[a][f] = __builtin_amdgcn_wmma_f32_16x16x32_f16(
                    false, af[a], false, bf, (short)0, acc[a][f], false, false);
        }
        __syncthreads();
    }

    // C layout: lane holds col = ln16, rows i + 8*hi16
#pragma unroll
    for (int a = 0; a < 2; ++a) {
        const int rbase = m0 + wm * 32 + a * 16 + hi16 * 8;
#pragma unroll
        for (int f = 0; f < 4; ++f) {
            int col = n0 + wn * 64 + f * 16 + ln16;
#pragma unroll
            for (int i = 0; i < 8; ++i)
                C[(size_t)(rbase + i) * N + col] = acc[a][f][i];
        }
    }
}

// ---------------------------------------------------------------------------
// Flash attention + top-K memory attention + gated fusion.
// grid = (B*H, T/128), block = 256 (8 waves, 16 q-rows per wave).
// qkv: [B,T,3E] f32; writes combined [B,T,E] f32.
// ---------------------------------------------------------------------------
__global__ __launch_bounds__(256) void flash_attn_mem(
    const float* __restrict__ qkv, const float* __restrict__ mem_k,
    const float* __restrict__ mem_v, const float* __restrict__ gate_bias,
    float* __restrict__ comb) {
    constexpr int T = 2048, E = 1024, H = 16, D = 64, E3 = 3 * E;

    __shared__ _Float16 Qs[128][72];     // [q][d]
    __shared__ _Float16 Ks[32][72];      // [key][d]
    __shared__ _Float16 Vs[64][40];      // [d][key] (transposed)
    __shared__ _Float16 Ps[8][16][40];   // per-wave P staging [q][key]

    const int tid  = threadIdx.x;
    const int lane = tid & 31;
    const int wave = tid >> 5;
    const int ln16 = lane & 15;
    const int hi16 = lane >> 4;
    const int bh = blockIdx.x;
    const int b  = bh / H;
    const int h  = bh % H;
    const int q0 = blockIdx.y * 128;
    const size_t qkv_bh = (size_t)b * T * E3 + (size_t)h * D;

    // stage Q tile 128x64 (f32 -> f16)
#pragma unroll
    for (int i = 0; i < 8; ++i) {
        int idx = tid + i * 256;            // 0..2047
        int r   = idx >> 4;                 // 16 float4 per row
        int c4  = (idx & 15) * 4;
        float4 f = *(const float4*)(qkv + qkv_bh + (size_t)(q0 + r) * E3 + c4);
        v4h p = {(_Float16)f.x, (_Float16)f.y, (_Float16)f.z, (_Float16)f.w};
        *(v4h*)&Qs[r][c4] = p;
    }

    float m_i[8], l_i[8];
    v8f o[4];
#pragma unroll
    for (int f = 0; f < 4; ++f)
#pragma unroll
        for (int i = 0; i < 8; ++i) o[f][i] = 0.0f;
#pragma unroll
    for (int i = 0; i < 8; ++i) { m_i[i] = NEG_MAX; l_i[i] = 0.0f; }

    const int hh = hi16 * 8;
    const int kb = hi16 * 16;
    const int ktend = q0 + 128;           // causal: keys <= q0+127

    for (int kt0 = 0; kt0 < ktend; kt0 += 32) {
        __syncthreads();                  // protect K/V tiles from prior iter
        // stage K (row-major) and V (transposed), f32 -> f16
#pragma unroll
        for (int i = 0; i < 2; ++i) {
            int idx = tid + i * 256;      // 0..511
            int kk  = idx >> 4;           // 16 float4 per key-row
            int c4  = (idx & 15) * 4;
            float4 f = *(const float4*)(qkv + qkv_bh + E +
                                        (size_t)(kt0 + kk) * E3 + c4);
            v4h p = {(_Float16)f.x, (_Float16)f.y, (_Float16)f.z, (_Float16)f.w};
            *(v4h*)&Ks[kk][c4] = p;
            float4 g = *(const float4*)(qkv + qkv_bh + 2 * E +
                                        (size_t)(kt0 + kk) * E3 + c4);
            Vs[c4 + 0][kk] = (_Float16)g.x;
            Vs[c4 + 1][kk] = (_Float16)g.y;
            Vs[c4 + 2][kk] = (_Float16)g.z;
            Vs[c4 + 3][kk] = (_Float16)g.w;
        }
        // prefetch next K/V tile one iteration ahead (global_prefetch_b8)
        if (kt0 + 32 < ktend && tid < 64) {
            int kk = tid >> 1;
            int half = (tid & 1) * 32;
            const float* pk = qkv + qkv_bh + E +
                              (size_t)(kt0 + 32 + kk) * E3 + half;
            const float* pv = qkv + qkv_bh + 2 * E +
                              (size_t)(kt0 + 32 + kk) * E3 + half;
            __builtin_prefetch(pk, 0, 3);
            __builtin_prefetch(pv, 0, 3);
        }
        __syncthreads();

        // S = Q * K^T  (16x32 per wave, two k-steps of 32 over D=64)
        v8f s[2];
#pragma unroll
        for (int f = 0; f < 2; ++f)
#pragma unroll
            for (int i = 0; i < 8; ++i) s[f][i] = 0.0f;
        const int qrow = wave * 16 + ln16;
#pragma unroll
        for (int ks = 0; ks < 2; ++ks) {
            v16h aq = cat8(*(const v8h*)&Qs[qrow][ks * 32 + hh],
                           *(const v8h*)&Qs[qrow][ks * 32 + 16 + hh]);
#pragma unroll
            for (int f = 0; f < 2; ++f) {
                int col = f * 16 + ln16;
                v16h bk = cat8(*(const v8h*)&Ks[col][ks * 32 + hh],
                               *(const v8h*)&Ks[col][ks * 32 + 16 + hh]);
                s[f] = __builtin_amdgcn_wmma_f32_16x16x32_f16(
                    false, aq, false, bk, (short)0, s[f], false, false);
            }
        }

        // online softmax on C-layout fragments (row = i + 8*hi16, col = ln16)
        const float scale = 0.125f;       // 1/sqrt(64)
        const int rowb = q0 + wave * 16 + hi16 * 8;
        const int col0 = kt0 + ln16;
#pragma unroll
        for (int i = 0; i < 8; ++i) {
            int row = rowb + i;
            float v0 = s[0][i] * scale;
            float v1 = s[1][i] * scale;
            if (col0 > row)      v0 = NEG_MAX;
            if (col0 + 16 > row) v1 = NEG_MAX;
            float rm = fmaxf(v0, v1);
            rm = fmaxf(rm, __shfl_xor(rm, 1, 32));
            rm = fmaxf(rm, __shfl_xor(rm, 2, 32));
            rm = fmaxf(rm, __shfl_xor(rm, 4, 32));
            rm = fmaxf(rm, __shfl_xor(rm, 8, 32));
            float mnew = fmaxf(m_i[i], rm);
            float corr = __expf(m_i[i] - mnew);
            float p0 = __expf(v0 - mnew);
            float p1 = __expf(v1 - mnew);
            float rs = p0 + p1;
            rs += __shfl_xor(rs, 1, 32);
            rs += __shfl_xor(rs, 2, 32);
            rs += __shfl_xor(rs, 4, 32);
            rs += __shfl_xor(rs, 8, 32);
            l_i[i] = l_i[i] * corr + rs;
            m_i[i] = mnew;
#pragma unroll
            for (int f = 0; f < 4; ++f) o[f][i] *= corr;
            Ps[wave][i + hi16 * 8][ln16]      = (_Float16)p0;
            Ps[wave][i + hi16 * 8][16 + ln16] = (_Float16)p1;
        }
        // in-wave LDS is serviced in order; fence compiler + HW explicitly
        asm volatile("s_wait_dscnt 0" ::: "memory");

        // O += P * V  (A-layout P from per-wave LDS, B from transposed Vs)
        v16h pf = cat8(*(const v8h*)&Ps[wave][ln16][hh],
                       *(const v8h*)&Ps[wave][ln16][16 + hh]);
#pragma unroll
        for (int f = 0; f < 4; ++f) {
            int dcol = f * 16 + ln16;
            v16h vf = cat8(*(const v8h*)&Vs[dcol][hh],
                           *(const v8h*)&Vs[dcol][16 + hh]);
            o[f] = __builtin_amdgcn_wmma_f32_16x16x32_f16(
                false, pf, false, vf, (short)0, o[f], false, false);
        }
    }

    // epilogue: 1/l normalization + K=3 memory attention + gated fusion
    const float g = gate_bias[h];
    for (int i = 0; i < 8; ++i) {
        int t = q0 + wave * 16 + hi16 * 8 + i;
        const float* qrow  = qkv + qkv_bh + (size_t)t * E3;
        const float* mkrow = mem_k + ((size_t)(b * T + t) * 3) * E + h * D;
        const float* mvrow = mem_v + ((size_t)(b * T + t) * 3) * E + h * D;
        float d0 = 0.f, d1 = 0.f, d2 = 0.f;
#pragma unroll
        for (int f = 0; f < 4; ++f) {
            int d = f * 16 + ln16;
            float qv = qrow[d];
            d0 += qv * mkrow[d];
            d1 += qv * mkrow[E + d];
            d2 += qv * mkrow[2 * E + d];
        }
        d0 += __shfl_xor(d0, 1, 32); d0 += __shfl_xor(d0, 2, 32);
        d0 += __shfl_xor(d0, 4, 32); d0 += __shfl_xor(d0, 8, 32);
        d1 += __shfl_xor(d1, 1, 32); d1 += __shfl_xor(d1, 2, 32);
        d1 += __shfl_xor(d1, 4, 32); d1 += __shfl_xor(d1, 8, 32);
        d2 += __shfl_xor(d2, 1, 32); d2 += __shfl_xor(d2, 2, 32);
        d2 += __shfl_xor(d2, 4, 32); d2 += __shfl_xor(d2, 8, 32);
        const float msc = 4096.0f;        // E * sqrt(H), per reference
        float mx = fmaxf(d0, fmaxf(d1, d2));
        float w0 = __expf((d0 - mx) * msc);
        float w1 = __expf((d1 - mx) * msc);
        float w2 = __expf((d2 - mx) * msc);
        float inv = 1.0f / (w0 + w1 + w2);
        w0 *= inv; w1 *= inv; w2 *= inv;
        float linv = 1.0f / l_i[i];
#pragma unroll
        for (int f = 0; f < 4; ++f) {
            int d = f * 16 + ln16;
            float mvv = w0 * mvrow[d] + w1 * mvrow[E + d] + w2 * mvrow[2 * E + d];
            float yv  = o[f][i] * linv;
            comb[(size_t)(b * T + t) * E + h * D + d] = mvv * g + yv * (1.0f - g);
        }
    }
}

// ---------------------------------------------------------------------------
extern "C" void kernel_launch(void* const* d_in, const int* in_sizes, int n_in,
                              void* d_out, int out_size, void* d_ws, size_t ws_size,
                              hipStream_t stream) {
    (void)in_sizes; (void)n_in; (void)out_size; (void)ws_size;
    constexpr int B = 4, T = 2048, E = 1024, H = 16;
    constexpr int MT = B * T;            // 8192 rows

    const float* x      = (const float*)d_in[0];
    const float* mem_k  = (const float*)d_in[1];
    const float* mem_v  = (const float*)d_in[2];
    const float* W_attn = (const float*)d_in[3];
    const float* W_proj = (const float*)d_in[4];
    const float* gate   = (const float*)d_in[5];
    float* out = (float*)d_out;

    float* qkv  = (float*)d_ws;                       // 8192 x 3072 f32 (96 MB)
    float* comb = qkv + (size_t)MT * 3 * E;           // 8192 x 1024 f32 (32 MB)

    // 1) qkv = x @ W_attn
    dim3 g1(MT / 128, (3 * E) / 128);
    gemm_f16_wmma<<<g1, 256, 0, stream>>>(x, W_attn, qkv, MT, 3 * E, E);

    // 2) flash attention + memory attention + gating -> combined
    dim3 g2(B * H, T / 128);
    flash_attn_mem<<<g2, 256, 0, stream>>>(qkv, mem_k, mem_v, gate, comb);

    // 3) out = combined @ W_proj
    dim3 g3(MT / 128, E / 128);
    gemm_f16_wmma<<<g3, 256, 0, stream>>>(comb, W_proj, out, MT, E, E);
}
